// SwinTransformerBlock_64355789964115
// MI455X (gfx1250) — compile-verified, matrix operand order checked by reading, not run on hardware
//
#include <hip/hip_runtime.h>

typedef __attribute__((ext_vector_type(16))) __bf16 v16bf;
typedef __attribute__((ext_vector_type(8)))  float  v8f;
typedef __attribute__((ext_vector_type(4)))  int    v4i;

__device__ __forceinline__ v8f wmma_bf16(v16bf a, v16bf b, v8f c) {
  return __builtin_amdgcn_wmma_f32_16x16x32_bf16(false, a, false, b, (short)0, c, false, false);
}

union FragU { uint4 q[2]; v16bf f; };

// Load a 16x32 bf16 A/B fragment (row-major source, ld elements per row).
// lane 0-15: row=row0+lane, K = k0+{0..7, 16..23}
// lane 16-31: row=row0+lane-16, K = k0+{8..15, 24..31}
__device__ __forceinline__ v16bf load_frag(const __bf16* base, int ld, int row0, int k0, int lane) {
  int hi = (lane >> 4) & 1;
  const __bf16* p = base + (size_t)(row0 + (lane & 15)) * ld + (k0 + hi * 8);
  FragU u;
  u.q[0] = *reinterpret_cast<const uint4*>(p);
  u.q[1] = *reinterpret_cast<const uint4*>(p + 16);
  return u.f;
}

// ---- gfx1250 async global->LDS copy (16B per lane), ASYNCcnt-tracked ----
#if defined(__has_builtin)
#if __has_builtin(__builtin_amdgcn_global_load_async_to_lds_b128) && \
    __has_builtin(__builtin_amdgcn_s_wait_asynccnt)
#define HAVE_ASYNC_LDS 1
#endif
#endif

typedef __attribute__((address_space(1))) v4i* gv4i_p;  // global int4*
typedef __attribute__((address_space(3))) v4i* lv4i_p;  // LDS int4*

__device__ __forceinline__ void async_copy16(const __bf16* g, __bf16* l) {
#ifdef HAVE_ASYNC_LDS
  __builtin_amdgcn_global_load_async_to_lds_b128((gv4i_p)(g), (lv4i_p)(l), 0, 0);
#else
  *reinterpret_cast<uint4*>(l) = *reinterpret_cast<const uint4*>(g);
#endif
}

#ifdef HAVE_ASYNC_LDS
#define WAIT_ASYNC(n) __builtin_amdgcn_s_wait_asynccnt(n)
#else
#define WAIT_ASYNC(n)
#endif

__device__ __forceinline__ float gelu_exact(float v) {
  return 0.5f * v * (1.f + erff(v * 0.7071067811865475f));
}

__device__ __forceinline__ void wave_red2(float& s, float& q) {
#pragma unroll
  for (int o = 16; o > 0; o >>= 1) {
    s += __shfl_xor(s, o, 32);
    q += __shfl_xor(q, o, 32);
  }
}

// ---------------- weight f32 -> bf16 ----------------
__global__ void k_cvt(const float* __restrict__ s, __bf16* __restrict__ d, int n) {
  int i = blockIdx.x * 256 + threadIdx.x;
  if (i < n) d[i] = (__bf16)s[i];
}

// ---------------- LN1 + shifted window partition (pad 49->64 rows) ----------------
// win[w][r][c], w = b*64 + wh*8 + wwv, r = i*7+j, source = x[b][(wh*7+i+3)%56][(wwv*7+j+3)%56][c]
__global__ __launch_bounds__(256) void k_ln1_win(
    const float* __restrict__ x, const float* __restrict__ g, const float* __restrict__ bb,
    __bf16* __restrict__ win) {
  int gw = blockIdx.x * 8 + (threadIdx.x >> 5);
  int lane = threadIdx.x & 31;
  __bf16* dst = win + (size_t)gw * 192;
  int r = gw & 63;
  if (r >= 49) {
#pragma unroll
    for (int k = 0; k < 6; ++k) dst[lane + 32 * k] = (__bf16)0.f;
    return;
  }
  int wid = gw >> 6;
  int b = wid >> 6, wi = wid & 63, wh = wi >> 3, wwv = wi & 7;
  int i = r / 7, j = r % 7;
  int ys = (wh * 7 + i + 3) % 56;
  int xs = (wwv * 7 + j + 3) % 56;
  const float* src = x + ((size_t)b * 3136 + ys * 56 + xs) * 192;
  float v[6], s = 0.f, q = 0.f;
#pragma unroll
  for (int k = 0; k < 6; ++k) { v[k] = src[lane + 32 * k]; s += v[k]; q += v[k] * v[k]; }
  wave_red2(s, q);
  float mean = s * (1.f / 192.f);
  float var = q * (1.f / 192.f) - mean * mean;
  float rstd = rsqrtf(var + 1e-5f);
#pragma unroll
  for (int k = 0; k < 6; ++k) {
    int c = lane + 32 * k;
    dst[c] = (__bf16)((v[k] - mean) * rstd * g[c] + bb[c]);
  }
}

// ---------------- LN2 (token-major) ----------------
__global__ __launch_bounds__(256) void k_ln2(
    const float* __restrict__ xin, const float* __restrict__ g, const float* __restrict__ bb,
    __bf16* __restrict__ out) {
  int gw = blockIdx.x * 8 + (threadIdx.x >> 5);
  int lane = threadIdx.x & 31;
  const float* src = xin + (size_t)gw * 192;
  __bf16* dst = out + (size_t)gw * 192;
  float v[6], s = 0.f, q = 0.f;
#pragma unroll
  for (int k = 0; k < 6; ++k) { v[k] = src[lane + 32 * k]; s += v[k]; q += v[k] * v[k]; }
  wave_red2(s, q);
  float mean = s * (1.f / 192.f);
  float var = q * (1.f / 192.f) - mean * mean;
  float rstd = rsqrtf(var + 1e-5f);
#pragma unroll
  for (int k = 0; k < 6; ++k) {
    int c = lane + 32 * k;
    dst[c] = (__bf16)((v[k] - mean) * rstd * g[c] + bb[c]);
  }
}

// ---------------- generic WMMA GEMM: out = A(MxK) @ Bw(NxK)^T ----------------
// 256 threads, block tile 128x64 (8 waves, 4x2 of 32x32). A/B tiles double-buffered
// through LDS with async global->LDS copies (ASYNCcnt), fragments via ds_load_b128.
enum { EP_BF16 = 0, EP_GELU = 1, EP_PROJ = 2, EP_FC2 = 3 };

template <int EPI>
__global__ __launch_bounds__(256) void k_gemm(
    const __bf16* __restrict__ A, int lda,
    const __bf16* __restrict__ Bw, int ldb,
    __bf16* __restrict__ outB, float* __restrict__ outF, int ldc,
    const float* __restrict__ bias, const float* __restrict__ shortcut, int K) {
  __shared__ __align__(16) __bf16 aS[2][128][32];  // 16 KB
  __shared__ __align__(16) __bf16 bS[2][64][32];   //  8 KB
  int tid = threadIdx.x;
  int lane = tid & 31;
  int wv = tid >> 5;
  int mBlk = blockIdx.x * 128;
  int nBlk = blockIdx.y * 64;
  int mw = (wv >> 1) * 32;   // wave row offset inside block tile
  int nw = (wv & 1) * 32;    // wave col offset inside block tile
  int m0 = mBlk + mw;
  int n0 = nBlk + nw;

  // 768 16B chunks per stage (A: 512, B: 256) -> 3 per thread
  auto issue_stage = [&](int st, int kt) {
#pragma unroll
    for (int u = 0; u < 3; ++u) {
      int c = tid + u * 256;
      if (c < 512) {
        int row = c >> 2, kc = c & 3;
        async_copy16(A + (size_t)(mBlk + row) * lda + kt * 32 + kc * 8,
                     &aS[st][row][kc * 8]);
      } else {
        int c2 = c - 512;
        int row = c2 >> 2, kc = c2 & 3;
        async_copy16(Bw + (size_t)(nBlk + row) * ldb + kt * 32 + kc * 8,
                     &bS[st][row][kc * 8]);
      }
    }
  };

  int KT = K >> 5;
  v8f acc[2][2] = {};
  issue_stage(0, 0);
  for (int kt = 0; kt < KT; ++kt) {
    int st = kt & 1;
    if (kt + 1 < KT) {
      issue_stage(st ^ 1, kt + 1);  // prefetch next stage
      WAIT_ASYNC(3);                // current stage's 3 copies complete (in-order)
    } else {
      WAIT_ASYNC(0);
    }
    __syncthreads();
    const __bf16* ap = &aS[st][0][0];
    const __bf16* bp = &bS[st][0][0];
    v16bf a0 = load_frag(ap, 32, mw, 0, lane);
    v16bf a1 = load_frag(ap, 32, mw + 16, 0, lane);
    v16bf b0 = load_frag(bp, 32, nw, 0, lane);
    v16bf b1 = load_frag(bp, 32, nw + 16, 0, lane);
    acc[0][0] = wmma_bf16(a0, b0, acc[0][0]);
    acc[0][1] = wmma_bf16(a0, b1, acc[0][1]);
    acc[1][0] = wmma_bf16(a1, b0, acc[1][0]);
    acc[1][1] = wmma_bf16(a1, b1, acc[1][1]);
    __syncthreads();
  }

  int hi = lane >> 4, nl = lane & 15;
#pragma unroll
  for (int ti = 0; ti < 2; ++ti)
#pragma unroll
    for (int tj = 0; tj < 2; ++tj) {
      int n = n0 + tj * 16 + nl;
#pragma unroll
      for (int r = 0; r < 8; ++r) {
        int m = m0 + ti * 16 + r + hi * 8;   // D layout: vgpr r -> M=r (+8 for upper lanes)
        float v = acc[ti][tj][r];
        if (EPI == EP_BF16) {
          outB[(size_t)m * ldc + n] = (__bf16)v;
        } else if (EPI == EP_GELU) {
          v = gelu_exact(v + bias[n]);
          outB[(size_t)m * ldc + n] = (__bf16)v;
        } else if (EPI == EP_PROJ) {
          v += bias[n];
          int wid = m >> 6, rr = m & 63;
          if (rr < 49) {  // skip pad rows; reverse window partition + roll(+3) + residual
            int b = wid >> 6, wi = wid & 63, wh = wi >> 3, wwv = wi & 7;
            int i = rr / 7, j = rr % 7;
            int y = (wh * 7 + i + 3) % 56;
            int x = (wwv * 7 + j + 3) % 56;
            size_t idx = ((size_t)b * 3136 + y * 56 + x) * 192 + n;
            outF[idx] = shortcut[idx] + v;
          }
        } else {  // EP_FC2: bias + gelu (faithful quirk) + residual into d_out
          v = gelu_exact(v + bias[n]);
          size_t idx = (size_t)m * (size_t)ldc + n;
          outF[idx] = outF[idx] + v;
        }
      }
    }
}

// ---------------- windowed attention: one wave per (window, head) ----------------
__global__ __launch_bounds__(256) void k_attn(
    const __bf16* __restrict__ qkv, __bf16* __restrict__ ao,
    const int* __restrict__ relc, const float* __restrict__ relb,
    const unsigned char* __restrict__ amask) {
  __shared__ __align__(16) float  sm[8][16][64];    // 32 KB: score strip per wave
  __shared__ __align__(16) __bf16 pbuf[8][16][64];  // 16 KB: softmaxed probs (bf16)
  int lane = threadIdx.x & 31;
  int wv = threadIdx.x >> 5;
  int g = blockIdx.x * 8 + wv;           // window-head id, 0..12287
  int w = g / 6, h = g - w * 6;
  int hi = lane >> 4, nl = lane & 15;
  const __bf16* base = qkv + (size_t)w * 64 * 576;
  const __bf16* qp = base + h * 32;
  const __bf16* kp = base + 192 + h * 32;
  const __bf16* vp = base + 384 + h * 32;

  // K fragments: S = Q @ K^T, B frag lane n holds row (n0+n) of K  -> same loader as A
  v16bf bk[4];
#pragma unroll
  for (int nt = 0; nt < 4; ++nt) bk[nt] = load_frag(kp, 576, nt * 16, 0, lane);

  // V fragments for O = P @ V: gather columns of V (lane n = feature d)
  v16bf vf[2][2];
#pragma unroll
  for (int ks = 0; ks < 2; ++ks)
#pragma unroll
    for (int nt = 0; nt < 2; ++nt) {
      v16bf f = {};
#pragma unroll
      for (int e = 0; e < 16; ++e) {
        int kk = ks * 32 + hi * 8 + ((e < 8) ? e : (e + 8));
        f[e] = vp[(size_t)kk * 576 + nt * 16 + nl];
      }
      vf[ks][nt] = f;
    }

  const float SCALE = 13.856406460551018f;  // sqrt(192) faithful quirk
  int mi = w & 63;                          // mask window index (Nw=64)
  for (int mt = 0; mt < 4; ++mt) {
    v16bf aq = load_frag(qp, 576, mt * 16, 0, lane);
    v8f s4[4] = {};
#pragma unroll
    for (int nt = 0; nt < 4; ++nt) s4[nt] = wmma_bf16(aq, bk[nt], s4[nt]);
    // scale + relative bias + shift mask into LDS; pad rows/cols -> -1e30
#pragma unroll
    for (int nt = 0; nt < 4; ++nt) {
      int j = nt * 16 + nl;
#pragma unroll
      for (int rr = 0; rr < 8; ++rr) {
        int li = rr + hi * 8;
        int iG = mt * 16 + li;
        float sv;
        if (iG < 49 && j < 49) {
          sv = s4[nt][rr] * SCALE + relb[relc[iG * 49 + j] * 6 + h];
          if (amask[(size_t)mi * 2401 + iG * 49 + j]) sv = -100.f;
        } else {
          sv = -1e30f;
        }
        sm[wv][li][j] = sv;
      }
    }
    __syncthreads();
    // row softmax (lane<16 owns one row of the strip), write bf16 probs
    if (lane < 16) {
      float* rp = &sm[wv][lane][0];
      float mx = -1e30f;
      for (int j = 0; j < 64; ++j) mx = fmaxf(mx, rp[j]);
      float sum = 0.f;
      for (int j = 0; j < 64; ++j) sum += __expf(rp[j] - mx);
      float inv = 1.f / sum;
      for (int j = 0; j < 64; ++j) pbuf[wv][lane][j] = (__bf16)(__expf(rp[j] - mx) * inv);
    }
    __syncthreads();
    // O strip = P(16x64) @ V(64x32), K depth 64 = two 32-steps
    const __bf16* pb = &pbuf[wv][0][0];
    v16bf ap0 = load_frag(pb, 64, 0, 0, lane);
    v16bf ap1 = load_frag(pb, 64, 0, 32, lane);
    v8f o0 = {}, o1 = {};
    o0 = wmma_bf16(ap0, vf[0][0], o0);
    o0 = wmma_bf16(ap1, vf[1][0], o0);
    o1 = wmma_bf16(ap0, vf[0][1], o1);
    o1 = wmma_bf16(ap1, vf[1][1], o1);
    __bf16* aop = ao + ((size_t)w * 64 + mt * 16) * 192 + h * 32;
#pragma unroll
    for (int rr = 0; rr < 8; ++rr) {
      int mrow = rr + hi * 8;
      aop[(size_t)mrow * 192 + nl] = (__bf16)o0[rr];
      aop[(size_t)mrow * 192 + 16 + nl] = (__bf16)o1[rr];
    }
    __syncthreads();
  }
}

extern "C" void kernel_launch(void* const* d_in, const int* in_sizes, int n_in,
                              void* d_out, int out_size, void* d_ws, size_t ws_size,
                              hipStream_t stream) {
  const float* x            = (const float*)d_in[0];
  const unsigned char* amsk = (const unsigned char*)d_in[3];
  const int*   relc         = (const int*)d_in[4];
  const float* n1w = (const float*)d_in[5];
  const float* n1b = (const float*)d_in[6];
  const float* qkv_w = (const float*)d_in[7];
  const float* relb  = (const float*)d_in[8];
  const float* proj_w = (const float*)d_in[9];
  const float* proj_b = (const float*)d_in[10];
  const float* n2w = (const float*)d_in[11];
  const float* n2b = (const float*)d_in[12];
  const float* fc1_w = (const float*)d_in[13];
  const float* fc1_b = (const float*)d_in[14];
  const float* fc2_w = (const float*)d_in[15];
  const float* fc2_b = (const float*)d_in[16];
  float* out = (float*)d_out;

  char* ws = (char*)d_ws;
  size_t off = 0;
  auto alloc = [&](size_t bytes) {
    off = (off + 255) & ~(size_t)255;
    size_t o = off; off += bytes; return o;
  };
  __bf16* wq  = (__bf16*)(ws + alloc((size_t)576 * 192 * 2));
  __bf16* wp  = (__bf16*)(ws + alloc((size_t)192 * 192 * 2));
  __bf16* w1  = (__bf16*)(ws + alloc((size_t)768 * 192 * 2));
  __bf16* w2  = (__bf16*)(ws + alloc((size_t)192 * 768 * 2));
  __bf16* bufA = (__bf16*)(ws + alloc((size_t)2048 * 64 * 192 * 2));  // WIN -> AO -> X2
  __bf16* bufB = (__bf16*)(ws + alloc((size_t)100352 * 768 * 2));     // QKV -> H1

  // 0) weights -> bf16
  k_cvt<<<(110592 + 255) / 256, 256, 0, stream>>>(qkv_w, wq, 110592);
  k_cvt<<<(36864 + 255) / 256, 256, 0, stream>>>(proj_w, wp, 36864);
  k_cvt<<<(147456 + 255) / 256, 256, 0, stream>>>(fc1_w, w1, 147456);
  k_cvt<<<(147456 + 255) / 256, 256, 0, stream>>>(fc2_w, w2, 147456);
  // 1) LN1 + shifted window partition: 2048*64 rows
  k_ln1_win<<<16384, 256, 0, stream>>>(x, n1w, n1b, bufA);
  // 2) QKV GEMM: (131072 x 192) @ (192 x 576)
  k_gemm<EP_BF16><<<dim3(1024, 9), 256, 0, stream>>>(bufA, 192, wq, 192, bufB, nullptr, 576,
                                                     nullptr, nullptr, 192);
  // 3) windowed attention -> AO (reuses WIN buffer)
  k_attn<<<1536, 256, 0, stream>>>(bufB, bufA, relc, relb, amsk);
  // 4) proj GEMM + bias + window-reverse + roll + residual -> d_out (x1)
  k_gemm<EP_PROJ><<<dim3(1024, 3), 256, 0, stream>>>(bufA, 192, wp, 192, nullptr, out, 192,
                                                     proj_b, x, 192);
  // 5) LN2 over x1 -> X2 (reuses AO buffer)
  k_ln2<<<12544, 256, 0, stream>>>(out, n2w, n2b, bufA);
  // 6) fc1 GEMM + bias + GELU: (100352 x 192) @ (192 x 768) -> H1 (reuses QKV buffer)
  k_gemm<EP_GELU><<<dim3(784, 12), 256, 0, stream>>>(bufA, 192, w1, 192, bufB, nullptr, 768,
                                                     fc1_b, nullptr, 192);
  // 7) fc2 GEMM + bias + GELU (quirk) + residual -> d_out
  k_gemm<EP_FC2><<<dim3(784, 3), 256, 0, stream>>>(bufB, 768, w2, 768, nullptr, out, 192,
                                                   fc2_b, nullptr, 768);
}
